// CrossGraphEncoder_15436112462316
// MI455X (gfx1250) — matrix operand order repaired and checked by reading, not compiled, not run on hardware
//
#include <hip/hip_runtime.h>

typedef unsigned int u32;
typedef unsigned short u16;
typedef __attribute__((ext_vector_type(16))) __bf16 v16bf;
typedef __attribute__((ext_vector_type(8)))  float  v8f;
typedef __attribute__((ext_vector_type(4)))  unsigned int v4u;

#define NB 4          // graphs
#define NA 256        // atoms per graph
#define NGRID 1728    // grid nodes per graph
#define BA (NB*NA)            // 1024
#define NNODES (BA + NB*NGRID) // 7936
#define KA 8
#define KG 32
#define EA (BA*KA)            // 8192 atom edges
#define ETOT (EA + NB*NGRID*KG) // 229376
#define CODE 64
#define HID 128
#define NW 4          // waves per block in layer kernel

union FragBF { v4u q[2]; v16bf v; };

__device__ __forceinline__ u16 f2bf(float f) {
  union { float f; u32 u; } x; x.f = f;
  u32 r = x.u + 0x7FFFu + ((x.u >> 16) & 1u);
  return (u16)(r >> 16);
}

#define LDS_SYNC() asm volatile("s_wait_dscnt 0" ::: "memory")

// ---------------- init: node_pos, h0 (one-hot / zeros), hb0 ----------------
__global__ void init_pos_h(const float* __restrict__ pos, const int* __restrict__ types,
                           float* __restrict__ npos, float* __restrict__ h0,
                           u16* __restrict__ hb0) {
  int n = blockIdx.x * 256 + threadIdx.x;
  if (n >= NNODES) return;
  if (n < BA) {
    npos[n*3+0] = pos[n*3+0]; npos[n*3+1] = pos[n*3+1]; npos[n*3+2] = pos[n*3+2];
    int t = types[n];
    for (int c = 0; c < CODE; ++c) {
      float v = (c == t) ? 1.0f : 0.0f;
      h0[n*CODE+c] = v; hb0[n*CODE+c] = f2bf(v);
    }
  } else {
    int gi = n - BA;
    int gl = gi % NGRID;
    int ix = gl / 144, iy = (gl / 12) % 12, iz = gl % 12;
    npos[n*3+0] = -8.25f + 1.5f * ix;
    npos[n*3+1] = -8.25f + 1.5f * iy;
    npos[n*3+2] = -8.25f + 1.5f * iz;
    for (int c = 0; c < CODE; ++c) { h0[n*CODE+c] = 0.0f; hb0[n*CODE+c] = 0; }
  }
}

// ---------------- pack weights: fold eW into W1, fragment layouts ----------
// W1p layout per layer: [kt(5)][nt(8)][lane(32)][j(16)] bf16 ; K = 32kt + j + 16*(lane>>4), n = 16nt + (lane&15)
// W2p layout per layer: [kt(4)][nt(4)][lane(32)][j(16)] bf16
__global__ void pack_kernel(const float* __restrict__ edge_W, const float* __restrict__ edge_b,
                            const float* __restrict__ W1, const float* __restrict__ b1,
                            const float* __restrict__ W2,
                            u16* __restrict__ W1p, u16* __restrict__ W2p,
                            float* __restrict__ b1c) {
  int l = blockIdx.x, tid = threadIdx.x;
  const float* eW = edge_W + l*20*64;
  const float* eb = edge_b + l*64;
  const float* w1 = W1 + l*192*128;
  const float* w2 = W2 + l*128*64;
  u16* w1p = W1p + l*20480;
  u16* w2p = W2p + l*8192;
  for (int j = tid; j < 128; j += 256) {
    float s = b1[l*128 + j];
    for (int k = 0; k < 64; ++k) s += eb[k] * w1[(128+k)*128 + j];
    b1c[l*128 + j] = s;
  }
  for (int idx = tid; idx < 20480; idx += 256) {
    int j = idx & 15, lane = (idx >> 4) & 31, nt = (idx >> 9) & 7, kt = idx >> 12;
    int K = 32*kt + j + 16*(lane >> 4);
    int n = 16*nt + (lane & 15);
    float val;
    if (K < 128) val = w1[K*128 + n];
    else if (K < 148) {
      int g = K - 128; float s = 0.0f;
      for (int k = 0; k < 64; ++k) s += eW[g*64 + k] * w1[(128+k)*128 + n];
      val = s;
    } else val = 0.0f;
    w1p[idx] = f2bf(val);
  }
  for (int idx = tid; idx < 8192; idx += 256) {
    int j = idx & 15, lane = (idx >> 4) & 31, nt = (idx >> 9) & 3, kt = idx >> 11;
    int K = 32*kt + j + 16*(lane >> 4);
    int n = 16*nt + (lane & 15);
    w2p[idx] = f2bf(w2[K*64 + n]);
  }
}

// ---------------- kNN edge build ----------------
__global__ void atom_knn(const float* __restrict__ pos, int* __restrict__ src) {
  int ai = blockIdx.x * 256 + threadIdx.x;
  if (ai >= BA) return;
  int b = ai >> 8, il = ai & 255;
  float px = pos[ai*3], py = pos[ai*3+1], pz = pos[ai*3+2];
  float nd[KA]; int ni[KA];
  for (int k = 0; k < KA; ++k) { nd[k] = 1e30f; ni[k] = 0; }
  for (int j = 0; j < NA; ++j) {
    if (j == il) continue;
    int aj = b*NA + j;
    float dx = pos[aj*3]-px, dy = pos[aj*3+1]-py, dz = pos[aj*3+2]-pz;
    float d2 = dx*dx + dy*dy + dz*dz;
    if (d2 < nd[KA-1]) {
      int p = KA-1;
      while (p > 0 && d2 < nd[p-1]) { nd[p] = nd[p-1]; ni[p] = ni[p-1]; --p; }
      nd[p] = d2; ni[p] = j;
    }
  }
  for (int k = 0; k < KA; ++k) src[ai*KA + k] = b*NA + ni[k];
}

__global__ void grid_knn(const float* __restrict__ pos, int* __restrict__ src) {
  __shared__ float ap[NA*3];
  int b = blockIdx.y, t = threadIdx.x;
  ap[t*3+0] = pos[(b*NA+t)*3+0];
  ap[t*3+1] = pos[(b*NA+t)*3+1];
  ap[t*3+2] = pos[(b*NA+t)*3+2];
  __syncthreads();
  int gl = blockIdx.x * 256 + t;
  if (gl >= NGRID) return;
  int ix = gl/144, iy = (gl/12)%12, iz = gl%12;
  float px = -8.25f + 1.5f*ix, py = -8.25f + 1.5f*iy, pz = -8.25f + 1.5f*iz;
  float nd[KG]; int ni[KG];
  for (int k = 0; k < KG; ++k) { nd[k] = 1e30f; ni[k] = 0; }
  for (int j = 0; j < NA; ++j) {
    float dx = ap[j*3]-px, dy = ap[j*3+1]-py, dz = ap[j*3+2]-pz;
    float d2 = dx*dx + dy*dy + dz*dz;
    if (d2 < nd[KG-1]) {
      int p = KG-1;
      while (p > 0 && d2 < nd[p-1]) { nd[p] = nd[p-1]; ni[p] = ni[p-1]; --p; }
      nd[p] = d2; ni[p] = j;
    }
  }
  int base = EA + (b*NGRID + gl)*KG;
  for (int k = 0; k < KG; ++k) src[base + k] = b*NA + ni[k];
}

// ---------------- fused layer: edge MLP (WMMA) + aggregate + LN ------------
// one wave per destination node
__global__ __launch_bounds__(NW*32) void layer_kernel(
    const u16* __restrict__ hbIn, const float* __restrict__ hIn,
    float* __restrict__ hOut, u16* __restrict__ hbOut,
    const int* __restrict__ src, const float* __restrict__ npos,
    const u16* __restrict__ W1p, const u16* __restrict__ W2p,
    const float* __restrict__ b1c, const float* __restrict__ b2,
    const float* __restrict__ lnG, const float* __restrict__ lnB) {
  __shared__ u16 At[NW][16*160];   // per-wave A tile: 16 edges x 160 feats (bf16)
  __shared__ u16 Mt[NW][16*128];   // per-wave hidden tile: 16 x 128 (bf16)
  __shared__ float gOff[20], gCoef[20];

  int tid = threadIdx.x;
  if (tid < 20) {
    float kk = __logf(6.0f) / 19.0f;
    gOff[tid] = __expf(tid * kk) - 1.0f;
    int tp = (tid < 1) ? 1 : tid;
    float delta = __expf(tp * kk) - __expf((tp - 1) * kk);
    gCoef[tid] = -0.5f / (delta * delta);
  }
  __syncthreads();

  int lane = tid & 31, w = tid >> 5;
  int node = blockIdx.x * NW + w;
  if (node >= NNODES) return;
  int hi = lane >> 4, ln = lane & 15;

  bool isAtom = node < BA;
  int nE = isAtom ? KA : KG;
  int nT = isAtom ? 1 : 2;
  int eBase = isAtom ? node*KA : EA + (node - BA)*KG;

  float dpx = npos[node*3], dpy = npos[node*3+1], dpz = npos[node*3+2];
  u16* A = At[w];
  u16* M = Mt[w];
  float aggr[4] = {0.f, 0.f, 0.f, 0.f};

  for (int t = 0; t < nT; ++t) {
    int realRows = isAtom ? 8 : 16;
    // ---- build A tile in LDS: [h_src(64) | h_dst(64) | gauss(20) | pad(12)]
    for (int r = 0; r < 16; ++r) {
      bool valid = r < realRows;
      u32 v1 = 0, v2 = 0; u16 gv = 0;
      if (valid) {
        int s = src[eBase + t*16 + r];
        v1 = ((const u32*)hbIn)[s*32 + lane];
        v2 = ((const u32*)hbIn)[node*32 + lane];
        if (lane < 20) {
          float rx = npos[s*3]-dpx, ry = npos[s*3+1]-dpy, rz = npos[s*3+2]-dpz;
          float d = sqrtf(rx*rx + ry*ry + rz*rz);
          d = fminf(d, 5.0f);
          float dd = d - gOff[lane];
          gv = f2bf(__expf(gCoef[lane] * dd * dd));
        }
      }
      ((u32*)(A + r*160))[lane] = v1;
      ((u32*)(A + r*160))[32 + lane] = v2;
      A[r*160 + 128 + lane] = (lane < 20) ? gv : (u16)0;
    }
    LDS_SYNC();

    // ---- GEMM1: [16x160] @ W1c [160x128] -> relu -> Mt (bf16)
    FragBF a1[5];
    #pragma unroll
    for (int kt = 0; kt < 5; ++kt) {
      a1[kt].q[0] = *(const v4u*)(A + ln*160 + 32*kt + 16*hi);
      a1[kt].q[1] = *(const v4u*)(A + ln*160 + 32*kt + 16*hi + 8);
    }
    #pragma unroll
    for (int nt = 0; nt < 8; ++nt) {
      v8f acc = {0.f,0.f,0.f,0.f,0.f,0.f,0.f,0.f};
      #pragma unroll
      for (int kt = 0; kt < 5; ++kt) {
        FragBF bfr;
        const v4u* wp = (const v4u*)(W1p + ((kt*8 + nt)*32 + lane)*16);
        bfr.q[0] = wp[0]; bfr.q[1] = wp[1];
        acc = __builtin_amdgcn_wmma_f32_16x16x32_bf16(false, a1[kt].v, false, bfr.v,
                                                      (short)0, acc, false, false);
      }
      int col = nt*16 + ln;
      float bias = b1c[col];
      #pragma unroll
      for (int v = 0; v < 8; ++v) {
        float x = fmaxf(acc[v] + bias, 0.0f);
        M[(v + 8*hi)*128 + col] = f2bf(x);
      }
    }
    LDS_SYNC();

    // ---- GEMM2: [16x128] @ W2 [128x64]; reduce rows into aggr
    FragBF a2[4];
    #pragma unroll
    for (int kt = 0; kt < 4; ++kt) {
      a2[kt].q[0] = *(const v4u*)(M + ln*128 + 32*kt + 16*hi);
      a2[kt].q[1] = *(const v4u*)(M + ln*128 + 32*kt + 16*hi + 8);
    }
    #pragma unroll
    for (int nt = 0; nt < 4; ++nt) {
      v8f acc = {0.f,0.f,0.f,0.f,0.f,0.f,0.f,0.f};
      #pragma unroll
      for (int kt = 0; kt < 4; ++kt) {
        FragBF bfr;
        const v4u* wp = (const v4u*)(W2p + ((kt*4 + nt)*32 + lane)*16);
        bfr.q[0] = wp[0]; bfr.q[1] = wp[1];
        acc = __builtin_amdgcn_wmma_f32_16x16x32_bf16(false, a2[kt].v, false, bfr.v,
                                                      (short)0, acc, false, false);
      }
      float s = acc[0]+acc[1]+acc[2]+acc[3]+acc[4]+acc[5]+acc[6]+acc[7];
      if (isAtom && hi) s = 0.0f;   // padded rows 8..15 excluded
      aggr[nt] += s;
    }
    LDS_SYNC();
  }

  // ---- aggregate across row-halves, residual, layernorm
  float res[4];
  #pragma unroll
  for (int nt = 0; nt < 4; ++nt) {
    float tot = aggr[nt] + __shfl_xor(aggr[nt], 16, 32);
    int col = nt*16 + ln;
    res[nt] = hIn[node*CODE + col] + tot / (float)nE + b2[col];
  }
  float s1 = res[0]+res[1]+res[2]+res[3];
  float s2 = res[0]*res[0]+res[1]*res[1]+res[2]*res[2]+res[3]*res[3];
  #pragma unroll
  for (int d = 1; d < 16; d <<= 1) {
    s1 += __shfl_xor(s1, d, 32);
    s2 += __shfl_xor(s2, d, 32);
  }
  float mu = s1 * (1.0f/64.0f);
  float var = s2 * (1.0f/64.0f) - mu*mu;
  float inv = rsqrtf(var + 1e-5f);
  if (hi == 0) {
    #pragma unroll
    for (int nt = 0; nt < 4; ++nt) {
      int col = nt*16 + ln;
      float y = (res[nt] - mu) * inv * lnG[col] + lnB[col];
      hOut[node*CODE + col] = y;
      hbOut[node*CODE + col] = f2bf(y);
    }
  }
}

// ---------------- launch ----------------
extern "C" void kernel_launch(void* const* d_in, const int* in_sizes, int n_in,
                              void* d_out, int out_size, void* d_ws, size_t ws_size,
                              hipStream_t stream) {
  const float* pos    = (const float*)d_in[0];
  const int*   types  = (const int*)d_in[1];
  const float* edge_W = (const float*)d_in[3];
  const float* edge_b = (const float*)d_in[4];
  const float* W1     = (const float*)d_in[5];
  const float* b1     = (const float*)d_in[6];
  const float* W2     = (const float*)d_in[7];
  const float* b2     = (const float*)d_in[8];
  const float* ln_g   = (const float*)d_in[9];
  const float* ln_b   = (const float*)d_in[10];

  char* ws = (char*)d_ws;
  size_t off = 0;
  float* npos = (float*)(ws + off); off += (size_t)NNODES*3*4;      // 95232
  int*   src  = (int*)(ws + off);   off += (size_t)ETOT*4;          // edges
  float* h0   = (float*)(ws + off); off += (size_t)NNODES*CODE*4;
  float* h1   = (float*)(ws + off); off += (size_t)NNODES*CODE*4;
  u16*   hb0  = (u16*)(ws + off);   off += (size_t)NNODES*CODE*2;
  u16*   hb1  = (u16*)(ws + off);   off += (size_t)NNODES*CODE*2;
  u16*   W1p  = (u16*)(ws + off);   off += (size_t)4*20480*2;
  u16*   W2p  = (u16*)(ws + off);   off += (size_t)4*8192*2;
  float* b1c  = (float*)(ws + off); off += (size_t)4*128*4;

  init_pos_h<<<(NNODES + 255)/256, 256, 0, stream>>>(pos, types, npos, h0, hb0);
  pack_kernel<<<4, 256, 0, stream>>>(edge_W, edge_b, W1, b1, W2, W1p, W2p, b1c);
  atom_knn<<<(BA + 255)/256, 256, 0, stream>>>(pos, src);
  grid_knn<<<dim3((NGRID + 255)/256, NB), 256, 0, stream>>>(pos, src);

  float* hc = h0; float* hn = h1;
  u16* hbc = hb0; u16* hbn = hb1;
  int blocks = (NNODES + NW - 1) / NW;
  for (int l = 0; l < 4; ++l) {
    layer_kernel<<<blocks, NW*32, 0, stream>>>(
        hbc, hc, hn, hbn, src, npos,
        W1p + l*20480, W2p + l*8192, b1c + l*128,
        b2 + l*64, ln_g + l*64, ln_b + l*64);
    float* tf = hc; hc = hn; hn = tf;
    u16* tb = hbc; hbc = hbn; hbn = tb;
  }
  // final result is in hc; grid-node tail is contiguous
  hipMemcpyAsync(d_out, hc + (size_t)BA*CODE, (size_t)NB*NGRID*CODE*4,
                 hipMemcpyDeviceToDevice, stream);
}
